// WVP_se_36361193128454
// MI455X (gfx1250) — matrix operand order repaired and checked by reading, not compiled, not run on hardware
//
#include <hip/hip_runtime.h>

// ---------------------------------------------------------------------------
// WVP model for MI455X (gfx1250): all dense GEMMs via v_wmma_f32_16x16x32_f16
// (f16 inputs, f32 accumulate). Each wave computes a 16x64 output strip:
// one A fragment feeds 4 WMMAs (4x less A re-traffic, 4x wmma density).
// Fragment loads are branch-free float4 (global_load_b128/ds_load_b128);
// edge guards are value-selects (v_cndmask), never exec-mask branches.
// ---------------------------------------------------------------------------

#define BSZ 96
#define TQ  512
#define ISZ 512
#define HSZ 128
#define NHS 16
#define THH 8
#define NHT (BSZ*NHS)     // 1536
#define QK  12
#define MW  4
#define NEGV (-1e10f)

typedef __attribute__((ext_vector_type(16))) _Float16 v16h;
typedef __attribute__((ext_vector_type(8)))  float    v8f;

static __device__ __forceinline__ v8f wmma_f16(const v16h& a, const v16h& b, const v8f& c) {
  return __builtin_amdgcn_wmma_f32_16x16x32_f16(false, a, false, b, (short)0, c, false, false);
}

static __device__ __forceinline__ float sigm(float x) { return 1.0f / (1.0f + __expf(-x)); }

static __device__ __forceinline__ v16h cvt16(float4 x0, float4 x1, float4 x2, float4 x3) {
  v16h r;
  r[0]=(_Float16)x0.x;  r[1]=(_Float16)x0.y;  r[2]=(_Float16)x0.z;  r[3]=(_Float16)x0.w;
  r[4]=(_Float16)x1.x;  r[5]=(_Float16)x1.y;  r[6]=(_Float16)x1.z;  r[7]=(_Float16)x1.w;
  r[8]=(_Float16)x2.x;  r[9]=(_Float16)x2.y;  r[10]=(_Float16)x2.z; r[11]=(_Float16)x2.w;
  r[12]=(_Float16)x3.x; r[13]=(_Float16)x3.y; r[14]=(_Float16)x3.z; r[15]=(_Float16)x3.w;
  return r;
}

// A fragment (16x32 f16, ISA 7.12.2): lane half h holds K = kb+8h+0..7 and kb+16+8h+0..7
static __device__ __forceinline__ v16h load_a_frag(const float* __restrict__ Ar, int kb, int half) {
  const float* p = Ar + kb + (half << 3);
  return cvt16(*(const float4*)(p), *(const float4*)(p + 4),
               *(const float4*)(p + 16), *(const float4*)(p + 20));
}
// B fragment (32x16 f16): lane half h holds K = kb+8h+0..7 and kb+16+8h+0..7 of column n
static __device__ __forceinline__ v16h load_b_frag(const float* __restrict__ Wr, int kb, int half) {
  const float* p = Wr + kb + (half << 3);
  return cvt16(*(const float4*)(p), *(const float4*)(p + 4),
               *(const float4*)(p + 16), *(const float4*)(p + 20));
}
// Guarded variants: clamp index, select value -> cndmask, no exec divergence.
static __device__ __forceinline__ v16h load_a_guard(const float* __restrict__ Ar, int kb, int half, int K) {
  v16h a;
#pragma unroll
  for (int e = 0; e < 16; ++e) {
    int ka = kb + ((e >> 3) << 4) + (half << 3) + (e & 7);
    float v = Ar[ka < K ? ka : 0];
    a[e] = (_Float16)(ka < K ? v : 0.0f);
  }
  return a;
}
static __device__ __forceinline__ v16h load_b_guard(const float* __restrict__ Wr, int kb, int half, int K, bool nv) {
  v16h b;
#pragma unroll
  for (int e = 0; e < 16; ++e) {
    int kw = kb + ((e >> 3) << 4) + (half << 3) + (e & 7);
    float v = Wr[kw < K ? kw : 0];
    b[e] = (_Float16)((nv && kw < K) ? v : 0.0f);
  }
  return b;
}

// ---------------------------------------------------------------------------
// Generic GEMM: C[M,N] = A[M,K] @ W[N,K]^T + bias[N]. M must be multiple of 16.
// One 16x64 strip per wave (4 n-tiles share one A fragment), 8 waves/block.
// ---------------------------------------------------------------------------
__global__ __launch_bounds__(256)
void gemm_bias_wmma(const float* __restrict__ A, const float* __restrict__ W,
                    const float* __restrict__ bias, float* __restrict__ C,
                    int M, int N, int K, int ldc, int colOff)
{
  int wave = threadIdx.x >> 5;
  int lane = threadIdx.x & 31;
  int half = lane >> 4, l15 = lane & 15;
  int ntN = (N + 15) >> 4;
  int ntG = (ntN + 3) >> 2;                  // groups of 4 n-tiles
  long tile  = (long)blockIdx.x * 8 + wave;
  long total = (long)(M >> 4) * ntG;
  if (tile >= total) return;                 // wave-uniform
  int mt = (int)(tile / ntG), ng = (int)(tile % ntG);
  int m0 = mt << 4, nbase = ng << 6;
  const float* Ar = A + (size_t)(m0 + l15) * K;
  int nj[4]; bool nv[4]; const float* Wr[4];
#pragma unroll
  for (int j = 0; j < 4; ++j) {
    nj[j] = nbase + (j << 4) + l15;
    nv[j] = (nj[j] < N);
    Wr[j] = W + (size_t)(nv[j] ? nj[j] : 0) * K;
  }
  bool fullG = (nbase + 64 <= N);            // wave-uniform
  int Kf = K & ~31;
  v8f acc[4];
#pragma unroll
  for (int j = 0; j < 4; ++j) acc[j] = (v8f){0.f,0.f,0.f,0.f,0.f,0.f,0.f,0.f};
  if (fullG) {
    for (int kb = 0; kb < Kf; kb += 32) {
      __builtin_prefetch(Ar + kb + 64, 0, 1);     // global_prefetch_b8
      v16h a = load_a_frag(Ar, kb, half);
      acc[0] = wmma_f16(a, load_b_frag(Wr[0], kb, half), acc[0]);
      acc[1] = wmma_f16(a, load_b_frag(Wr[1], kb, half), acc[1]);
      acc[2] = wmma_f16(a, load_b_frag(Wr[2], kb, half), acc[2]);
      acc[3] = wmma_f16(a, load_b_frag(Wr[3], kb, half), acc[3]);
    }
    if (Kf < K) {
      v16h a = load_a_guard(Ar, Kf, half, K);
#pragma unroll
      for (int j = 0; j < 4; ++j)
        acc[j] = wmma_f16(a, load_b_guard(Wr[j], Kf, half, K, true), acc[j]);
    }
  } else {
    for (int kb = 0; kb < K; kb += 32) {
      v16h a = load_a_guard(Ar, kb, half, K);
#pragma unroll
      for (int j = 0; j < 4; ++j)
        acc[j] = wmma_f16(a, load_b_guard(Wr[j], kb, half, K, nv[j]), acc[j]);
    }
  }
#pragma unroll
  for (int j = 0; j < 4; ++j) {
    if (nv[j]) {
      float bn = bias[nj[j]];
#pragma unroll
      for (int r = 0; r < 8; ++r) {
        int mr = m0 + r + (half << 3);        // C rows r / r+8 per lane half
        C[(size_t)mr * ldc + colOff + nj[j]] = acc[j][r] + bn;
      }
    }
  }
}

// ---------------------------------------------------------------------------
// Persistent LSTM recurrence: 16 sequences/block, h/c in LDS.
// Per step: g[16,256] = h[16,64] @ Whh[256,64]^T via WMMA (A fragments loaded
// once per wave, reused for both n-tiles), then gates (i,f,g,o).
// ---------------------------------------------------------------------------
__global__ __launch_bounds__(256)
void lstm_rec_wmma(const float* __restrict__ xg, const float* __restrict__ Whh,
                   const int* __restrict__ lens, float* __restrict__ hs,
                   int rows, int T, int ldh, int colOff, int backward)
{
  __shared__ float sh[16 * 64];
  __shared__ float sc[16 * 64];
  __shared__ float sg[16 * 256];
  int tid = threadIdx.x;
  int wave = tid >> 5, lane = tid & 31;
  int half = lane >> 4, l15 = lane & 15;
  int r0 = blockIdx.x * 16;
  for (int i = tid; i < 16 * 64; i += 256) { sh[i] = 0.0f; sc[i] = 0.0f; }
  __syncthreads();
  for (int s = 0; s < T; ++s) {
    {
      const float* Ar = sh + l15 * 64;
      v16h a0 = load_a_frag(Ar, 0, half);
      v16h a1 = load_a_frag(Ar, 32, half);
#pragma unroll
      for (int j = 0; j < 2; ++j) {
        int n0 = ((wave << 1) + j) << 4;
        const float* Wr = Whh + (size_t)(n0 + l15) * 64;
        v8f acc = {0.f,0.f,0.f,0.f,0.f,0.f,0.f,0.f};
        acc = wmma_f16(a0, load_b_frag(Wr, 0, half),  acc);
        acc = wmma_f16(a1, load_b_frag(Wr, 32, half), acc);
#pragma unroll
        for (int r = 0; r < 8; ++r)
          sg[(r + (half << 3)) * 256 + n0 + l15] = acc[r];
      }
    }
    __syncthreads();
    for (int cell = tid; cell < 16 * 64; cell += 256) {
      int lr = cell >> 6, u = cell & 63;
      int gr = r0 + lr;
      if (gr < rows) {
        int L = lens[gr];
        if (s < L) {
          int t = backward ? (L - 1 - s) : s;
          const float* xr = xg + ((size_t)gr * T + t) * 256;
          float gi = xr[u]       + sg[lr * 256 + u];
          float gf = xr[64 + u]  + sg[lr * 256 + 64 + u];
          float gg = xr[128 + u] + sg[lr * 256 + 128 + u];
          float go = xr[192 + u] + sg[lr * 256 + 192 + u];
          float cn = sigm(gf) * sc[cell] + sigm(gi) * tanhf(gg);
          float hn = sigm(go) * tanhf(cn);
          sc[cell] = cn;
          sh[cell] = hn;
          hs[((size_t)gr * T + t) * ldh + colOff + u] = hn;
        }
      }
    }
    __syncthreads();
  }
}

// one-hot knowledge tail + zero padded question rows (prevents NaN leakage)
__global__ void wenc_tail_kernel(const int* __restrict__ know, const int* __restrict__ l_n,
                                 float* __restrict__ wenc)
{
  int idx = blockIdx.x * blockDim.x + threadIdx.x;
  if (idx >= BSZ * TQ) return;
  int b = idx / TQ, t = idx % TQ;
  float* row = wenc + (size_t)idx * 140;
  if (t >= l_n[b]) {
    for (int d = 0; d < 140; ++d) row[d] = 0.0f;
  } else {
    int k = know[idx];
    for (int q = 0; q < QK; ++q) row[128 + q] = (q == k) ? 1.0f : 0.0f;
  }
}

__global__ void wench_kernel(const float* __restrict__ hall, const int* __restrict__ l_hpu,
                             const int* __restrict__ khdr, float* __restrict__ wenc_hs)
{
  int idx = blockIdx.x * blockDim.x + threadIdx.x;
  if (idx >= NHT * 132) return;
  int nh = idx / 132, d = idx % 132;
  float v;
  if (d < 128) v = hall[((size_t)nh * THH + (l_hpu[nh] - 1)) * 128 + d];
  else         v = (khdr[nh] == (d - 128)) ? 1.0f : 0.0f;
  wenc_hs[idx] = v;
}

__global__ void hsob_kernel(const float* __restrict__ wenc_hs, const int* __restrict__ wc,
                            const int* __restrict__ wn, float* __restrict__ hs_ob)
{
  int idx = blockIdx.x * blockDim.x + threadIdx.x;
  if (idx >= BSZ * MW * 132) return;
  int d = idx % 132, bw = idx / 132;
  int b = bw >> 2, w = bw & 3;
  int col = (w < wn[b]) ? wc[bw] : 0;
  hs_ob[idx] = wenc_hs[((size_t)b * NHS + col) * 132 + d];
}

// attention scores + masked softmax + context, one block per (b,w)
__global__ __launch_bounds__(256)
void attn_kernel(const float* __restrict__ attx, const float* __restrict__ wenc,
                 const float* __restrict__ hs_ob, const int* __restrict__ l_n,
                 float* __restrict__ c_n)
{
  __shared__ float sq[132];
  __shared__ float sp[TQ];
  __shared__ float red[256];
  int b = blockIdx.x >> 2, w = blockIdx.x & 3;
  int tid = threadIdx.x;
  const float* qr = hs_ob + ((size_t)b * MW + w) * 132;
  if (tid < 132) sq[tid] = qr[tid];
  __syncthreads();
  int L = l_n[b];
  for (int t = tid; t < TQ; t += 256) {
    float v = NEGV;
    if (t < L) {
      const float4* ar = (const float4*)(attx + ((size_t)b * TQ + t) * 132);
      const float4* q4 = (const float4*)sq;
      float s = 0.0f;
#pragma unroll 3
      for (int d = 0; d < 33; ++d) {
        float4 a = ar[d], q = q4[d];
        s += a.x * q.x + a.y * q.y + a.z * q.z + a.w * q.w;
      }
      v = s;
    }
    sp[t] = v;
  }
  __syncthreads();
  float m = -1e30f;
  for (int t = tid; t < TQ; t += 256) m = fmaxf(m, sp[t]);
  red[tid] = m; __syncthreads();
  for (int st = 128; st > 0; st >>= 1) { if (tid < st) red[tid] = fmaxf(red[tid], red[tid + st]); __syncthreads(); }
  float mx = red[0]; __syncthreads();
  float sa = 0.0f;
  for (int t = tid; t < TQ; t += 256) { float e = __expf(sp[t] - mx); sp[t] = e; sa += e; }
  red[tid] = sa; __syncthreads();
  for (int st = 128; st > 0; st >>= 1) { if (tid < st) red[tid] += red[tid + st]; __syncthreads(); }
  float inv = 1.0f / red[0];
  __syncthreads();
  for (int d = tid; d < 140; d += 256) {
    float s = 0.0f;
    for (int t = 0; t < TQ; ++t) s += sp[t] * wenc[((size_t)b * TQ + t) * 140 + d];
    c_n[((size_t)b * MW + w) * 140 + d] = s * inv;
  }
}

// vec[b,w,384] = [ c_n@Wc^T+b | hs_ob@Whs^T+b | Wop[:,op]+b ]
__global__ void vec_kernel(const float* __restrict__ c_n, const float* __restrict__ hs_ob,
                           const float* __restrict__ Wc_w, const float* __restrict__ Wc_b,
                           const float* __restrict__ Whs_w, const float* __restrict__ Whs_b,
                           const float* __restrict__ Wop_w, const float* __restrict__ Wop_b,
                           const int* __restrict__ wn, const int* __restrict__ wo,
                           float* __restrict__ vec)
{
  int idx = blockIdx.x * blockDim.x + threadIdx.x;
  if (idx >= BSZ * MW * 384) return;
  int j = idx % 384, bw = idx / 384;
  int b = bw >> 2, w = bw & 3;
  float out;
  if (j < 128) {
    const float4* cr = (const float4*)(c_n + (size_t)bw * 140);
    const float4* wr = (const float4*)(Wc_w + (size_t)j * 140);
    float s = Wc_b[j];
    for (int d = 0; d < 35; ++d) {
      float4 a = cr[d], q = wr[d];
      s += a.x * q.x + a.y * q.y + a.z * q.z + a.w * q.w;
    }
    out = s;
  } else if (j < 256) {
    int jj = j - 128;
    const float4* hr = (const float4*)(hs_ob + (size_t)bw * 132);
    const float4* wr = (const float4*)(Whs_w + (size_t)jj * 132);
    float s = Whs_b[jj];
    for (int d = 0; d < 33; ++d) {
      float4 a = hr[d], q = wr[d];
      s += a.x * q.x + a.y * q.y + a.z * q.z + a.w * q.w;
    }
    out = s;
  } else {
    int jj = j - 256;
    int op = (w < wn[b]) ? wo[bw] : 0;
    out = Wop_w[jj * 4 + op] + Wop_b[jj];
  }
  vec[idx] = out;
}

__global__ void out_init_kernel(const float* __restrict__ b2, float* __restrict__ out)
{
  int idx = blockIdx.x * blockDim.x + threadIdx.x;
  if (idx >= BSZ * MW * TQ * 2) return;
  out[idx] = b2[idx & 1];
}

// fused out1 GEMM (virtual concat vec||wenc_n, K=524) + tanh + out2 fold-in.
// One wave = 16 rows x 64 cols (4 n-tiles, shared A fragment); out2 folded
// across the 4 tiles before the 16-lane shuffle reduction -> 2 atomics/row/wave.
__global__ __launch_bounds__(256)
void final_kernel(const float* __restrict__ vec, const float* __restrict__ wenc,
                  const float* __restrict__ W1, const float* __restrict__ b1,
                  const float* __restrict__ W2, float* __restrict__ out)
{
  const int Mtot = BSZ * MW * TQ;      // 196608
  const int Ktot = 524;
  int wave = threadIdx.x >> 5;
  int lane = threadIdx.x & 31;
  int half = lane >> 4, l15 = lane & 15;
  long tile  = (long)blockIdx.x * 8 + wave;
  long total = (long)(Mtot >> 4) * 2;  // 2 groups of 4 n-tiles (N=128)
  if (tile >= total) return;
  int mt = (int)(tile >> 1), g = (int)(tile & 1);
  int m0 = mt << 4, nbase = g << 6;
  int m = m0 + l15;
  int b = m / (MW * TQ);
  int rem = m - b * (MW * TQ);
  int w = rem >> 9, t = rem & 511;
  const float* pv = vec + ((size_t)b * MW + w) * 384;
  const float* pn = wenc + ((size_t)b * TQ + t) * 140;
  int nj[4]; const float* Wr[4];
#pragma unroll
  for (int j = 0; j < 4; ++j) {
    nj[j] = nbase + (j << 4) + l15;
    Wr[j] = W1 + (size_t)nj[j] * Ktot;
  }
  v8f acc[4];
#pragma unroll
  for (int j = 0; j < 4; ++j) acc[j] = (v8f){0.f,0.f,0.f,0.f,0.f,0.f,0.f,0.f};
  // region 1: vec part (K 0..383, 12 full chunks)
  for (int kb = 0; kb < 384; kb += 32) {
    v16h a = load_a_frag(pv, kb, half);
    acc[0] = wmma_f16(a, load_b_frag(Wr[0], kb, half), acc[0]);
    acc[1] = wmma_f16(a, load_b_frag(Wr[1], kb, half), acc[1]);
    acc[2] = wmma_f16(a, load_b_frag(Wr[2], kb, half), acc[2]);
    acc[3] = wmma_f16(a, load_b_frag(Wr[3], kb, half), acc[3]);
  }
  // region 2: wenc part (K 384..511, 4 full chunks)
  for (int kb = 384; kb < 512; kb += 32) {
    v16h a = load_a_frag(pn, kb - 384, half);
#pragma unroll
    for (int j = 0; j < 4; ++j)
      acc[j] = wmma_f16(a, load_b_frag(Wr[j], kb, half), acc[j]);
  }
  // region 3: remainder (K 512..523), guarded value-selects
  {
    v16h a = load_a_guard(pn, 128, half, 140);
#pragma unroll
    for (int j = 0; j < 4; ++j)
      acc[j] = wmma_f16(a, load_b_guard(Wr[j], 512, half, Ktot, true), acc[j]);
  }
  float bj[4], w20[4], w21[4];
#pragma unroll
  for (int j = 0; j < 4; ++j) { bj[j] = b1[nj[j]]; w20[j] = W2[nj[j]]; w21[j] = W2[128 + nj[j]]; }
#pragma unroll
  for (int r = 0; r < 8; ++r) {
    float p0 = 0.0f, p1 = 0.0f;
#pragma unroll
    for (int j = 0; j < 4; ++j) {
      float s1 = tanhf(acc[j][r] + bj[j]);
      p0 += s1 * w20[j];
      p1 += s1 * w21[j];
    }
#pragma unroll
    for (int msk = 1; msk < 16; msk <<= 1) {   // reduce within each 16-lane half
      p0 += __shfl_xor(p0, msk, 32);
      p1 += __shfl_xor(p1, msk, 32);
    }
    if (l15 == 0) {
      int mr = m0 + r + (half << 3);
      atomicAdd(&out[(size_t)mr * 2 + 0], p0);
      atomicAdd(&out[(size_t)mr * 2 + 1], p1);
    }
  }
}

__global__ void mask_kernel(const int* __restrict__ l_n, float* __restrict__ out)
{
  int idx = blockIdx.x * blockDim.x + threadIdx.x;
  if (idx >= BSZ * MW * TQ) return;
  int b = idx / (MW * TQ);
  int t = idx & 511;
  if (t >= l_n[b]) { out[(size_t)idx * 2] = NEGV; out[(size_t)idx * 2 + 1] = NEGV; }
}

// ---------------------------------------------------------------------------
extern "C" void kernel_launch(void* const* d_in, const int* in_sizes, int n_in,
                              void* d_out, int out_size, void* d_ws, size_t ws_size,
                              hipStream_t stream)
{
  (void)in_sizes; (void)n_in; (void)out_size; (void)ws_size;
  const float* wemb_n   = (const float*)d_in[0];
  const float* wemb_hpu = (const float*)d_in[1];
  const float* Wih_n0f = (const float*)d_in[2];  const float* Whh_n0f = (const float*)d_in[3];  const float* b_n0f = (const float*)d_in[4];
  const float* Wih_n0b = (const float*)d_in[5];  const float* Whh_n0b = (const float*)d_in[6];  const float* b_n0b = (const float*)d_in[7];
  const float* Wih_n1f = (const float*)d_in[8];  const float* Whh_n1f = (const float*)d_in[9];  const float* b_n1f = (const float*)d_in[10];
  const float* Wih_n1b = (const float*)d_in[11]; const float* Whh_n1b = (const float*)d_in[12]; const float* b_n1b = (const float*)d_in[13];
  const float* Wih_h0f = (const float*)d_in[14]; const float* Whh_h0f = (const float*)d_in[15]; const float* b_h0f = (const float*)d_in[16];
  const float* Wih_h0b = (const float*)d_in[17]; const float* Whh_h0b = (const float*)d_in[18]; const float* b_h0b = (const float*)d_in[19];
  const float* Wih_h1f = (const float*)d_in[20]; const float* Whh_h1f = (const float*)d_in[21]; const float* b_h1f = (const float*)d_in[22];
  const float* Wih_h1b = (const float*)d_in[23]; const float* Whh_h1b = (const float*)d_in[24]; const float* b_h1b = (const float*)d_in[25];
  const float* Watt_w = (const float*)d_in[26]; const float* Watt_b = (const float*)d_in[27];
  const float* Wc_w   = (const float*)d_in[28]; const float* Wc_b   = (const float*)d_in[29];
  const float* Whs_w  = (const float*)d_in[30]; const float* Whs_b  = (const float*)d_in[31];
  const float* Wop_w  = (const float*)d_in[32]; const float* Wop_b  = (const float*)d_in[33];
  const float* out1_w = (const float*)d_in[34]; const float* out1_b = (const float*)d_in[35];
  const float* out2_w = (const float*)d_in[36]; const float* out2_b = (const float*)d_in[37];
  const int* l_n   = (const int*)d_in[38];
  const int* l_hpu = (const int*)d_in[39];
  const int* wc    = (const int*)d_in[40];
  const int* wn    = (const int*)d_in[41];
  const int* wo    = (const int*)d_in[42];
  const int* know  = (const int*)d_in[43];
  const int* khdr  = (const int*)d_in[44];
  float* out = (float*)d_out;

  // workspace layout (floats) — peak ~105 MB with region reuse
  float* ws   = (float*)d_ws;
  float* XG   = ws;                          // 49152*256 (xg scratch; later attx)
  float* HS0  = ws + 12582912;               // 49152*128 (layer0 out; later header hs)
  float* WENC = HS0 + 6291456;               // 49152*140 (wenc_n)
  float* WHSB = WENC + 6881280;              // 1536*132
  float* HSOB = WHSB + 202752;               //  384*132
  float* CN   = HSOB + 50688;                //  384*140
  float* VEC  = CN + 53760;                  //  384*384
  float* HSH  = HS0;                         // 12288*128 (header layer0 out)
  float* HALL = HS0 + 12288 * 128;           // 12288*128 (header layer1 out)

  dim3 blk(256);
  auto gblocks = [](long M, long N) {
    long ntG = (((N + 15) / 16) + 3) / 4;
    long tiles = (M / 16) * ntG;
    return (int)((tiles + 7) / 8);
  };
  const int MQ = BSZ * TQ;     // 49152
  const int MH = NHT * THH;    // 12288

  // ---- question encoder ----
  gemm_bias_wmma<<<gblocks(MQ, 256), blk, 0, stream>>>(wemb_n, Wih_n0f, b_n0f, XG, MQ, 256, ISZ, 256, 0);
  lstm_rec_wmma<<<BSZ / 16, blk, 0, stream>>>(XG, Whh_n0f, l_n, HS0, BSZ, TQ, 128, 0, 0);
  gemm_bias_wmma<<<gblocks(MQ, 256), blk, 0, stream>>>(wemb_n, Wih_n0b, b_n0b, XG, MQ, 256, ISZ, 256, 0);
  lstm_rec_wmma<<<BSZ / 16, blk, 0, stream>>>(XG, Whh_n0b, l_n, HS0, BSZ, TQ, 128, 64, 1);
  gemm_bias_wmma<<<gblocks(MQ, 256), blk, 0, stream>>>(HS0, Wih_n1f, b_n1f, XG, MQ, 256, HSZ, 256, 0);
  lstm_rec_wmma<<<BSZ / 16, blk, 0, stream>>>(XG, Whh_n1f, l_n, WENC, BSZ, TQ, 140, 0, 0);
  gemm_bias_wmma<<<gblocks(MQ, 256), blk, 0, stream>>>(HS0, Wih_n1b, b_n1b, XG, MQ, 256, HSZ, 256, 0);
  lstm_rec_wmma<<<BSZ / 16, blk, 0, stream>>>(XG, Whh_n1b, l_n, WENC, BSZ, TQ, 140, 64, 1);
  wenc_tail_kernel<<<(MQ + 255) / 256, blk, 0, stream>>>(know, l_n, WENC);

  // ---- header encoder ----
  gemm_bias_wmma<<<gblocks(MH, 256), blk, 0, stream>>>(wemb_hpu, Wih_h0f, b_h0f, XG, MH, 256, ISZ, 256, 0);
  lstm_rec_wmma<<<NHT / 16, blk, 0, stream>>>(XG, Whh_h0f, l_hpu, HSH, NHT, THH, 128, 0, 0);
  gemm_bias_wmma<<<gblocks(MH, 256), blk, 0, stream>>>(wemb_hpu, Wih_h0b, b_h0b, XG, MH, 256, ISZ, 256, 0);
  lstm_rec_wmma<<<NHT / 16, blk, 0, stream>>>(XG, Whh_h0b, l_hpu, HSH, NHT, THH, 128, 64, 1);
  gemm_bias_wmma<<<gblocks(MH, 256), blk, 0, stream>>>(HSH, Wih_h1f, b_h1f, XG, MH, 256, HSZ, 256, 0);
  lstm_rec_wmma<<<NHT / 16, blk, 0, stream>>>(XG, Whh_h1f, l_hpu, HALL, NHT, THH, 128, 0, 0);
  gemm_bias_wmma<<<gblocks(MH, 256), blk, 0, stream>>>(HSH, Wih_h1b, b_h1b, XG, MH, 256, HSZ, 256, 0);
  lstm_rec_wmma<<<NHT / 16, blk, 0, stream>>>(XG, Whh_h1b, l_hpu, HALL, NHT, THH, 128, 64, 1);
  wench_kernel<<<(NHT * 132 + 255) / 256, blk, 0, stream>>>(HALL, l_hpu, khdr, WHSB);
  hsob_kernel<<<(BSZ * MW * 132 + 255) / 256, blk, 0, stream>>>(WHSB, wc, wn, HSOB);

  // ---- attention + condition vector ----
  gemm_bias_wmma<<<gblocks(MQ, 132), blk, 0, stream>>>(WENC, Watt_w, Watt_b, XG /*attx*/, MQ, 132, 140, 132, 0);
  attn_kernel<<<BSZ * MW, blk, 0, stream>>>(XG, WENC, HSOB, l_n, CN);
  vec_kernel<<<(BSZ * MW * 384 + 255) / 256, blk, 0, stream>>>(CN, HSOB, Wc_w, Wc_b, Whs_w, Whs_b,
                                                               Wop_w, Wop_b, wn, wo, VEC);

  // ---- fused output MLP ----
  out_init_kernel<<<(BSZ * MW * TQ * 2 + 255) / 256, blk, 0, stream>>>(out2_b, out);
  final_kernel<<<(BSZ * MW * TQ / 16) * 2 / 8, blk, 0, stream>>>(VEC, WENC, out1_w, out1_b, out2_w, out);
  mask_kernel<<<(BSZ * MW * TQ + 255) / 256, blk, 0, stream>>>(l_n, out);
}
